// dct_8x8_18021682774933
// MI455X (gfx1250) — matrix operand order repaired and checked by reading, not compiled, not run on hardware
//
#include <hip/hip_runtime.h>

typedef __attribute__((ext_vector_type(16))) _Float16 v16h;
typedef __attribute__((ext_vector_type(8)))  _Float16 v8h;
typedef __attribute__((ext_vector_type(8)))  float    v8f;
typedef __attribute__((ext_vector_type(4)))  float    v4f;

// G[k][n] for the fused DCT-II: k = x*8+y (pixel), n = u*8+v (frequency)
// G[k][n] = cos((2x+1)u*pi/16) * cos((2y+1)v*pi/16) * alpha(u)*alpha(v)/4
__device__ __forceinline__ float dct_g(int k, int n) {
    const float PI16 = 0.19634954084936207f; // pi/16
    int x = k >> 3, y = k & 7, u = n >> 3, v = n & 7;
    float cu = __cosf((float)((2 * x + 1) * u) * PI16);
    float cv = __cosf((float)((2 * y + 1) * v) * PI16);
    float s = 0.25f;
    if (u == 0) s *= 0.70710678118654752f;
    if (v == 0) s *= 0.70710678118654752f;
    return cu * cv * s;
}

__global__ __launch_bounds__(256) void dct8x8_wmma_kernel(
    const float* __restrict__ in, float* __restrict__ out,
    int n_rows, int n_tiles) {
    // Transposed basis table in LDS: Gt[n][k], so each lane's B-fragment
    // (two runs of 8 consecutive K for one N, 16 apart) are each contiguous -> ds_load_b128 x2.
    __shared__ _Float16 Gt[64 * 64];

    int tid = threadIdx.x;
    for (int i = tid; i < 64 * 64; i += 256) {
        int n = i >> 6, k = i & 63;
        Gt[i] = (_Float16)dct_g(k, n);
    }
    __syncthreads();

    int lane  = tid & 31;
    int gwave = blockIdx.x * (blockDim.x >> 5) + (tid >> 5);
    int nwave = gridDim.x * (blockDim.x >> 5);

    // --- B fragments: loop-invariant, live in VGPRs ---
    // Dense f16 B (32x16): element e<8 <-> K = kb+e, e>=8 <-> K = 16+kb+(e-8),
    // kb = 8*(lane>=16) (N=lane&15), matching the A fragment order.
    v16h B[2][4];
    {
        int nrow = lane & 15;
        int kb0  = (lane >> 4) * 8;
        #pragma unroll
        for (int t = 0; t < 2; ++t)
            #pragma unroll
            for (int c = 0; c < 4; ++c) {
                const _Float16* p = &Gt[(c * 16 + nrow) * 64 + t * 32 + kb0];
                union { v16h v; v8h h[2]; } ub;
                ub.h[0] = *(const v8h*)(p);
                ub.h[1] = *(const v8h*)(p + 16);
                B[t][c] = ub.v;
            }
    }

    for (int tile = gwave; tile < n_tiles; tile += nwave) {
        // --- A fragments straight from global in WMMA layout ---
        // A (16x32 f16): lane L -> row M=L&15; K covers
        //   {half..half+7} U {16+half..16+half+7},  half = 8*(L>=16)
        // i.e. four contiguous float4 loads per K-step, tile fully covered.
        int rowA = tile * 16 + (lane & 15);
        if (rowA >= n_rows) rowA = n_rows - 1; // clamp: EXEC stays all-ones for WMMA
        const float* rp = in + (size_t)rowA * 64;
        int half = (lane >> 4) * 8;

        // Uniform prefetch of next grid-stride tile (-> global_prefetch_b8):
        // pulls the next 4KB/wave tile toward L2 while this one computes.
        if (tile + nwave < n_tiles)
            __builtin_prefetch(rp + (size_t)nwave * 1024, 0, 1);

        v16h A[2];
        #pragma unroll
        for (int t = 0; t < 2; ++t) {
            v4f p0 = *(const v4f*)(rp + t * 32 + half);
            v4f p1 = *(const v4f*)(rp + t * 32 + half + 4);
            v4f p2 = *(const v4f*)(rp + t * 32 + 16 + half);
            v4f p3 = *(const v4f*)(rp + t * 32 + 16 + half + 4);
            union { v16h v; _Float16 e[16]; } ua;
            #pragma unroll
            for (int i = 0; i < 4; ++i) {
                ua.e[i]      = (_Float16)(p0[i] - 128.0f);
                ua.e[4 + i]  = (_Float16)(p1[i] - 128.0f);
                ua.e[8 + i]  = (_Float16)(p2[i] - 128.0f);
                ua.e[12 + i] = (_Float16)(p3[i] - 128.0f);
            }
            A[t] = ua.v;
        }

        // --- 8 WMMAs: 4 N-tiles x 2 K-steps ---
        v8f acc[4];
        #pragma unroll
        for (int c = 0; c < 4; ++c) {
            v8f z = {};
            acc[c] = __builtin_amdgcn_wmma_f32_16x16x32_f16(
                false, A[0], false, B[0][c], (short)0, z, false, false);
            acc[c] = __builtin_amdgcn_wmma_f32_16x16x32_f16(
                false, A[1], false, B[1][c], (short)0, acc[c], false, false);
        }

        // --- store: D VGPR r -> row rbase+r (lanes 0-15) / rbase+8+r (16-31) ---
        // Single base pointer; all 32 stores use immediate offsets so the
        // compiler emits one clause of global_store_b32 with no address math.
        int rbase = tile * 16 + (lane >> 4) * 8;
        float* op = out + (size_t)rbase * 64 + (lane & 15);
        if (tile * 16 + 16 <= n_rows) { // full tile: uniform branch, no guards
            #pragma unroll
            for (int r = 0; r < 8; ++r)
                #pragma unroll
                for (int c = 0; c < 4; ++c)
                    op[r * 64 + c * 16] = acc[c][r];
        } else { // ragged tail (never taken for 786432 rows; kept for safety)
            #pragma unroll
            for (int r = 0; r < 8; ++r)
                if (rbase + r < n_rows)
                    #pragma unroll
                    for (int c = 0; c < 4; ++c)
                        op[r * 64 + c * 16] = acc[c][r];
        }
    }
}

extern "C" void kernel_launch(void* const* d_in, const int* in_sizes, int n_in,
                              void* d_out, int out_size, void* d_ws, size_t ws_size,
                              hipStream_t stream) {
    const float* in  = (const float*)d_in[0];
    float*       out = (float*)d_out;

    long long n = (long long)in_sizes[0];      // 64*3*4096*8*8 = 50,331,648
    int n_rows  = (int)(n / 64);               // 786,432 blocks (rows of the GEMM)
    int n_tiles = (n_rows + 15) / 16;          // 49,152 M-tiles

    // Persistent-ish grid: amortize LDS table build + B-fragment setup
    // over ~3 tiles/wave while still saturating a large-WGP-count device.
    int blocks = (n_tiles + 7) / 8;
    if (blocks > 2048) blocks = 2048;
    if (blocks < 1) blocks = 1;

    dct8x8_wmma_kernel<<<blocks, 256, 0, stream>>>(in, out, n_rows, n_tiles);
}
